// ChromaSingleStreamBlock_63737314673116
// MI455X (gfx1250) — compile-verified, matrix operand order checked by reading, not run on hardware
//
#include <hip/hip_runtime.h>
#include <hip/hip_bf16.h>

// ---------------------------------------------------------------- constants
#define H_   24
#define HD_  128
#define DIM_ 3072
#define MLP_ 12288
#define L_   2048
#define O1_  (3*DIM_ + MLP_)     // 21504
#define C2K_ (DIM_ + MLP_)       // 15360
#define EPS_ 1e-6f
#define ATT_SCALE 0.08838834764831845f   // 128^-0.5

typedef __attribute__((ext_vector_type(16))) __bf16 bf16x16;
typedef __attribute__((ext_vector_type(2)))  __bf16 bf16x2;
typedef __attribute__((ext_vector_type(8)))  float  f32x8;
typedef __attribute__((ext_vector_type(2)))  float  f32x2;

// ----------------------------------------------------- async-to-LDS helpers
#if defined(__has_builtin)
#if __has_builtin(__builtin_amdgcn_global_load_async_to_lds_b128) && \
    __has_builtin(__builtin_amdgcn_s_wait_asynccnt)
#define HAVE_ASYNC_LDS 1
#endif
#endif
#ifndef HAVE_ASYNC_LDS
#define HAVE_ASYNC_LDS 0
#endif

#if HAVE_ASYNC_LDS
typedef int v4i_ __attribute__((vector_size(16)));
typedef __attribute__((address_space(1))) v4i_ gv4i_t;   // global 4xi32
typedef __attribute__((address_space(3))) v4i_ lv4i_t;   // LDS    4xi32
#endif

__device__ inline void async_copy_b128(const void* g, void* l) {
#if HAVE_ASYNC_LDS
    // AS(1) cast keeps full 64-bit address; AS(3) cast keeps the low 32 bits,
    // which per the ISA aperture rule is exactly the LDS byte offset.
    __builtin_amdgcn_global_load_async_to_lds_b128(
        (gv4i_t*)(uintptr_t)(g),
        (lv4i_t*)(uintptr_t)(l),
        0, 0);
#else
    *(uint4*)l = *(const uint4*)g;
#endif
}
__device__ inline void async_wait0() {
#if HAVE_ASYNC_LDS
    __builtin_amdgcn_s_wait_asynccnt(0);
#endif
}

// ------------------------------------------------------------- bf16 helpers
// Native converts: let the compiler emit v_cvt_(pk_)bf16_f32 instead of
// scalarized bit-twiddling.
__device__ inline __bf16 f2bf(float f) { return (__bf16)f; }
__device__ inline unsigned pk2(float lo, float hi) {
    union { bf16x2 b; unsigned u; } o;
    f32x2 f; f[0] = lo; f[1] = hi;
    o.b = __builtin_convertvector(f, bf16x2);
    return o.u;
}

// Fragment loader: element e of lane L covers K = 16*(e>=8) + 8*(L>=16) + (e&7)
// within a 32-wide K window.  Caller passes p = row_base + k_window + (lane>=16 ? 8 : 0).
__device__ inline bf16x16 load_frag(const __bf16* p) {
    union { uint4 q[2]; bf16x16 v; } u;
    u.q[0] = *(const uint4*)(p);        // elements 0..7   (K +0..7)
    u.q[1] = *(const uint4*)(p + 16);   // elements 8..15  (K +16..23)
    return u.v;
}

__device__ inline f32x8 wmma_bf16(bf16x16 a, bf16x16 b, f32x8 c) {
    return __builtin_amdgcn_wmma_f32_16x16x32_bf16(false, a, false, b,
                                                   (short)0, c, false, false);
}

__device__ inline float half16_max(float x) {
    x = fmaxf(x, __shfl_xor(x, 1, 32));
    x = fmaxf(x, __shfl_xor(x, 2, 32));
    x = fmaxf(x, __shfl_xor(x, 4, 32));
    x = fmaxf(x, __shfl_xor(x, 8, 32));
    return x;
}
__device__ inline float half16_sum(float x) {
    x += __shfl_xor(x, 1, 32);
    x += __shfl_xor(x, 2, 32);
    x += __shfl_xor(x, 4, 32);
    x += __shfl_xor(x, 8, 32);
    return x;
}
__device__ inline float gelu_tanh(float v) {
    float t = tanhf(0.7978845608028654f * (v + 0.044715f * v * v * v));
    return 0.5f * v * (1.f + t);
}

// ------------------------------------------------- Kernel 1: LayerNorm + mod
__global__ __launch_bounds__(256, 1)
void ln_mod_kernel(const float* __restrict__ X, const float* __restrict__ msc,
                   const float* __restrict__ msh, __bf16* __restrict__ Xm) {
    __shared__ float red[256];
    const int l = blockIdx.x, tid = threadIdx.x;
    const float* xr = X + (size_t)l * DIM_;

    float s = 0.f;
    for (int c = tid; c < DIM_; c += 256) s += xr[c];
    red[tid] = s; __syncthreads();
    for (int o = 128; o > 0; o >>= 1) { if (tid < o) red[tid] += red[tid + o]; __syncthreads(); }
    const float mu = red[0] * (1.f / DIM_);
    __syncthreads();

    float s2 = 0.f;
    for (int c = tid; c < DIM_; c += 256) { float d = xr[c] - mu; s2 += d * d; }
    red[tid] = s2; __syncthreads();
    for (int o = 128; o > 0; o >>= 1) { if (tid < o) red[tid] += red[tid + o]; __syncthreads(); }
    const float rs = rsqrtf(red[0] * (1.f / DIM_) + EPS_);

    for (int c = tid; c < DIM_; c += 256) {
        float ln = (xr[c] - mu) * rs;
        Xm[(size_t)l * DIM_ + c] = f2bf((1.f + msc[c]) * ln + msh[c]);
    }
}

// ---------------------------------------------------------- GEMM mainloop
// Block tile 128x128, 8 waves (2 M x 4 N), wave tile 64x32, BK=32.
// Double-buffered LDS: A staged by async-to-LDS b128, B (fp32->bf16) pipelined
// through registers.  One barrier per K-step.
#define STG 40                 // LDS row stride (bf16), keeps 16B alignment
#define TILE_E (128 * STG)     // elems per stage buffer

__device__ inline void gemm_mainloop(const __bf16* __restrict__ A, int lda,
                                     const float* __restrict__ B, int ldb,
                                     int K, int m0, int n0,
                                     __bf16* At, __bf16* Bt,    // 2 buffers each
                                     f32x8 (&acc)[4][2]) {
    const int tid  = threadIdx.x;
    const int lane = tid & 31, wid = tid >> 5;
    const int wm = wid >> 2, wn = wid & 3;
    const int ln15 = lane & 15, koff = (lane >> 4) * 8;
    const int sr = tid >> 1, sh = (tid & 1) * 16;

    const __bf16* aptr = A + (size_t)(m0 + sr) * lda + sh;
    const float*  bptr = B + (size_t)(n0 + sr) * ldb + sh;
    const int nk = K >> 5;

    // ---- prologue: stage tile 0 into buffer 0
    async_copy_b128(aptr,     At + sr * STG + sh);
    async_copy_b128(aptr + 8, At + sr * STG + sh + 8);
    {
        float4 f0 = ((const float4*)bptr)[0];
        float4 f1 = ((const float4*)bptr)[1];
        float4 f2 = ((const float4*)bptr)[2];
        float4 f3 = ((const float4*)bptr)[3];
        uint4 p0, p1;
        p0.x = pk2(f0.x, f0.y); p0.y = pk2(f0.z, f0.w);
        p0.z = pk2(f1.x, f1.y); p0.w = pk2(f1.z, f1.w);
        p1.x = pk2(f2.x, f2.y); p1.y = pk2(f2.z, f2.w);
        p1.z = pk2(f3.x, f3.y); p1.w = pk2(f3.z, f3.w);
        *(uint4*)(Bt + sr * STG + sh)     = p0;
        *(uint4*)(Bt + sr * STG + sh + 8) = p1;
    }
    async_wait0();
    __syncthreads();

    int cur = 0;
    for (int kt = 0; kt < nk; ++kt) {
        const int nxt = cur ^ 1;
        __bf16* Ac = At + cur * TILE_E;
        __bf16* Bc = Bt + cur * TILE_E;
        const bool more = (kt + 1 < nk);

        float4 f0, f1, f2, f3;
        if (more) {
            // issue next A tile (async -> LDS, overlaps WMMAs below)
            const __bf16* an = aptr + (size_t)(kt + 1) * 32;
            async_copy_b128(an,     At + nxt * TILE_E + sr * STG + sh);
            async_copy_b128(an + 8, At + nxt * TILE_E + sr * STG + sh + 8);
            // issue next B tile loads (latency hidden by WMMAs)
            const float* bn = bptr + (size_t)(kt + 1) * 32;
            f0 = ((const float4*)bn)[0];
            f1 = ((const float4*)bn)[1];
            f2 = ((const float4*)bn)[2];
            f3 = ((const float4*)bn)[3];
        }

        bf16x16 af[4], bfr[2];
#pragma unroll
        for (int ms = 0; ms < 4; ++ms)
            af[ms] = load_frag(Ac + (wm * 64 + ms * 16 + ln15) * STG + koff);
#pragma unroll
        for (int ns = 0; ns < 2; ++ns)
            bfr[ns] = load_frag(Bc + (wn * 32 + ns * 16 + ln15) * STG + koff);
#pragma unroll
        for (int ms = 0; ms < 4; ++ms)
#pragma unroll
            for (int ns = 0; ns < 2; ++ns)
                acc[ms][ns] = wmma_bf16(af[ms], bfr[ns], acc[ms][ns]);

        if (more) {
            uint4 p0, p1;
            p0.x = pk2(f0.x, f0.y); p0.y = pk2(f0.z, f0.w);
            p0.z = pk2(f1.x, f1.y); p0.w = pk2(f1.z, f1.w);
            p1.x = pk2(f2.x, f2.y); p1.y = pk2(f2.z, f2.w);
            p1.z = pk2(f3.x, f3.y); p1.w = pk2(f3.z, f3.w);
            *(uint4*)(Bt + nxt * TILE_E + sr * STG + sh)     = p0;
            *(uint4*)(Bt + nxt * TILE_E + sr * STG + sh + 8) = p1;
        }
        async_wait0();
        __syncthreads();
        cur = nxt;
    }
}

// ---------------------------------- Kernel 2: GEMM1 + qkv-norm/rope/v/gelu
#define TP 132   // fp32 epilogue tile row stride

union Smem1 {
    struct { __bf16 At[2 * TILE_E]; __bf16 Bt[2 * TILE_E]; } s;
    float tile[128 * TP];
};

__global__ __launch_bounds__(256, 1)
void gemm1_kernel(const __bf16* __restrict__ Xm, const float* __restrict__ W1,
                  const float* __restrict__ b1,  const float* __restrict__ pe,
                  const float* __restrict__ qnw, const float* __restrict__ knw,
                  __bf16* __restrict__ Qb, __bf16* __restrict__ Kb,
                  __bf16* __restrict__ Vb, __bf16* __restrict__ Comb) {
    __shared__ Smem1 sm;
    const int mb = blockIdx.x & 15, nb = blockIdx.x >> 4;   // 16 M x 168 N
    const int m0 = mb * 128, n0 = nb * 128;
    const int tid = threadIdx.x, lane = tid & 31, wid = tid >> 5;
    const int wm = wid >> 2, wn = wid & 3;
    const int ln15 = lane & 15, hi = lane >> 4;

    f32x8 acc[4][2];
#pragma unroll
    for (int ms = 0; ms < 4; ++ms)
#pragma unroll
        for (int ns = 0; ns < 2; ++ns)
#pragma unroll
            for (int v = 0; v < 8; ++v) acc[ms][ns][v] = 0.f;

    gemm_mainloop(Xm, DIM_, W1, DIM_, DIM_, m0, n0, sm.s.At, sm.s.Bt, acc);

    // spill accumulators to fp32 LDS tile (stage buffers are dead)
#pragma unroll
    for (int ms = 0; ms < 4; ++ms)
#pragma unroll
        for (int ns = 0; ns < 2; ++ns)
#pragma unroll
            for (int v = 0; v < 8; ++v) {
                int r = wm * 64 + ms * 16 + v + 8 * hi;
                int c = wn * 32 + ns * 16 + ln15;
                sm.tile[r * TP + c] = acc[ms][ns][v];
            }
    __syncthreads();

    // 256 threads: thread pair (2r, 2r+1) shares output row r; each does 64 cols
    const int row = tid >> 1, half = tid & 1, hb = half * 64;
    const int l = m0 + row;
    float vals[64];
    float ss = 0.f;
#pragma unroll
    for (int j = 0; j < 64; ++j) {
        float v = sm.tile[row * TP + hb + j] + b1[n0 + hb + j];
        vals[j] = v;
        ss += v * v;
    }

    if (nb < 48) {                       // ---- q (nb<24) or k (24..47)
        float tot = ss + __shfl_xor(ss, 1, 32);
        float rms = rsqrtf(tot * (1.f / HD_) + EPS_);
        const float* nw = (nb < 24) ? qnw : knw;
        __bf16* dst = (nb < 24) ? Qb : Kb;
        const int hh = nb % 24;
        const float sgn = half ? 1.f : -1.f;    // rotate_half sign
#pragma unroll
        for (int j = 0; j < 64; ++j) {
            int d = hb + j;
            float nv = vals[j] * rms * nw[d];
            float pv = __shfl_xor(nv, 1, 32);   // partner half's value at d^64
            float fr = pe[(size_t)l * HD_ + d];
            float o  = nv * cosf(fr) + sgn * pv * sinf(fr);
            dst[((size_t)hh * L_ + l) * HD_ + d] = f2bf(o);
        }
    } else if (nb < 72) {                // ---- v : store transposed [h][hd][l]
        const int hh = nb - 48;
#pragma unroll
        for (int j = 0; j < 64; ++j)
            Vb[((size_t)hh * HD_ + hb + j) * L_ + l] = f2bf(vals[j]);
    } else {                             // ---- mlp : gelu into combined
        const int c0 = DIM_ + (nb - 72) * 128 + hb;
#pragma unroll
        for (int j = 0; j < 64; ++j)
            Comb[(size_t)l * C2K_ + c0 + j] = f2bf(gelu_tanh(vals[j]));
    }
}

// ------------------------------------------- Kernel 3: flash attention (bf16)
#define QKS 136   // Q/K LDS row stride (bf16)
#define VPS 72    // V / P LDS row stride (bf16)

__global__ __launch_bounds__(128, 1)
void attn_kernel(const __bf16* __restrict__ Qb, const __bf16* __restrict__ Kb,
                 const __bf16* __restrict__ Vb, const float* __restrict__ mask,
                 __bf16* __restrict__ Comb) {
    __shared__ __bf16 Qs[64 * QKS];
    __shared__ __bf16 Ks[2][64 * QKS];
    __shared__ __bf16 Vs[2][128 * VPS];
    __shared__ __bf16 Ps[4 * 16 * VPS];

    const int h = blockIdx.x >> 5, qb = blockIdx.x & 31;
    const int tid = threadIdx.x, lane = tid & 31, wid = tid >> 5;
    const int ln15 = lane & 15, hi = lane >> 4, koff = hi * 8;
    const int qwave = qb * 64 + wid * 16;

    const __bf16* Qg = Qb + ((size_t)h * L_ + qb * 64) * HD_;
    const __bf16* Kg = Kb + (size_t)h * L_ * HD_;
    const __bf16* Vg = Vb + (size_t)h * HD_ * L_;

    const int r2 = tid >> 1, ch = (tid & 1) * 64;

    // ---- prologue: stage Q tile and K/V tile 0 (async)
#pragma unroll
    for (int i = 0; i < 8; ++i)
        async_copy_b128(Qg + (size_t)r2 * HD_ + ch + i * 8, Qs + r2 * QKS + ch + i * 8);
#pragma unroll
    for (int i = 0; i < 8; ++i)
        async_copy_b128(Kg + (size_t)r2 * HD_ + ch + i * 8, Ks[0] + r2 * QKS + ch + i * 8);
#pragma unroll
    for (int i = 0; i < 8; ++i)
        async_copy_b128(Vg + (size_t)tid * L_ + i * 8, Vs[0] + tid * VPS + i * 8);
    async_wait0();
    __syncthreads();

    bf16x16 qf[4];
#pragma unroll
    for (int kk = 0; kk < 4; ++kk)
        qf[kk] = load_frag(Qs + (wid * 16 + ln15) * QKS + kk * 32 + koff);

    f32x8 Of[8];
    float m_i[8], l_i[8];
#pragma unroll
    for (int ns = 0; ns < 8; ++ns)
#pragma unroll
        for (int v = 0; v < 8; ++v) Of[ns][v] = 0.f;
#pragma unroll
    for (int v = 0; v < 8; ++v) { m_i[v] = -1e30f; l_i[v] = 0.f; }

    int cur = 0;
    for (int kb = 0; kb < 32; ++kb) {
        const int nxt = cur ^ 1;
        if (kb + 1 < 32) {   // stage next K/V tiles while computing
#pragma unroll
            for (int i = 0; i < 8; ++i)
                async_copy_b128(Kg + (size_t)((kb + 1) * 64 + r2) * HD_ + ch + i * 8,
                                Ks[nxt] + r2 * QKS + ch + i * 8);
#pragma unroll
            for (int i = 0; i < 8; ++i)
                async_copy_b128(Vg + (size_t)tid * L_ + (kb + 1) * 64 + i * 8,
                                Vs[nxt] + tid * VPS + i * 8);
        }

        // ---- S = scale * Q K^T + mask  (16 queries x 64 keys per wave)
        f32x8 sv[4];
#pragma unroll
        for (int ns = 0; ns < 4; ++ns) {
            f32x8 s;
#pragma unroll
            for (int v = 0; v < 8; ++v) s[v] = 0.f;
#pragma unroll
            for (int kk = 0; kk < 4; ++kk)
                s = wmma_bf16(qf[kk],
                              load_frag(Ks[cur] + (ns * 16 + ln15) * QKS + kk * 32 + koff), s);
#pragma unroll
            for (int v = 0; v < 8; ++v) {
                size_t mi = (size_t)(qwave + v + 8 * hi) * L_ + kb * 64 + ns * 16 + ln15;
                s[v] = s[v] * ATT_SCALE + mask[mi];
            }
            sv[ns] = s;
        }

        // ---- online softmax update (rows r = v + 8*hi)
        float alpha[8];
#pragma unroll
        for (int v = 0; v < 8; ++v) {
            float mx = fmaxf(fmaxf(sv[0][v], sv[1][v]), fmaxf(sv[2][v], sv[3][v]));
            mx = half16_max(mx);
            float mn = fmaxf(m_i[v], mx);
            alpha[v] = __expf(m_i[v] - mn);
            float rs = 0.f;
#pragma unroll
            for (int ns = 0; ns < 4; ++ns) {
                float p = __expf(sv[ns][v] - mn);
                sv[ns][v] = p;
                rs += p;
            }
            rs = half16_sum(rs);
            l_i[v] = l_i[v] * alpha[v] + rs;
            m_i[v] = mn;
        }
#pragma unroll
        for (int ns = 0; ns < 8; ++ns)
#pragma unroll
            for (int v = 0; v < 8; ++v) Of[ns][v] *= alpha[v];

        // ---- C-layout -> A-layout transpose of P via per-wave LDS scratch
        __bf16* pw = Ps + wid * (16 * VPS);
#pragma unroll
        for (int ns = 0; ns < 4; ++ns)
#pragma unroll
            for (int v = 0; v < 8; ++v)
                pw[(v + 8 * hi) * VPS + ns * 16 + ln15] = f2bf(sv[ns][v]);
        // same-wave LDS RAW: lockstep issue + compiler dscnt wait, no barrier needed
        bf16x16 pf[2];
#pragma unroll
        for (int kk = 0; kk < 2; ++kk)
            pf[kk] = load_frag(pw + ln15 * VPS + kk * 32 + koff);

        // ---- O += P V   (contraction over 64 keys = 2 k-steps)
#pragma unroll
        for (int ns = 0; ns < 8; ++ns)
#pragma unroll
            for (int kk = 0; kk < 2; ++kk)
                Of[ns] = wmma_bf16(pf[kk],
                                   load_frag(Vs[cur] + (ns * 16 + ln15) * VPS + kk * 32 + koff),
                                   Of[ns]);

        async_wait0();       // next K/V tiles landed in LDS
        __syncthreads();     // all waves done reading cur, writes visible
        cur = nxt;
    }

    // ---- normalize and write attn into combined[:, 0:3072]
#pragma unroll
    for (int ns = 0; ns < 8; ++ns)
#pragma unroll
        for (int v = 0; v < 8; ++v) {
            float o = Of[ns][v] / l_i[v];
            size_t idx = (size_t)(qwave + v + 8 * hi) * C2K_ + h * HD_ + ns * 16 + ln15;
            Comb[idx] = f2bf(o);
        }
}

// -------------------------------- Kernel 4: GEMM2 + bias + gate + residual
__global__ __launch_bounds__(256, 1)
void gemm2_kernel(const __bf16* __restrict__ Comb, const float* __restrict__ W2,
                  const float* __restrict__ b2,   const float* __restrict__ X,
                  const float* __restrict__ gate, float* __restrict__ Out) {
    __shared__ struct { __bf16 At[2 * TILE_E]; __bf16 Bt[2 * TILE_E]; } sm;
    const int mb = blockIdx.x & 15, nb = blockIdx.x >> 4;   // 16 M x 24 N
    const int m0 = mb * 128, n0 = nb * 128;
    const int tid = threadIdx.x, lane = tid & 31, wid = tid >> 5;
    const int wm = wid >> 2, wn = wid & 3;
    const int ln15 = lane & 15, hi = lane >> 4;

    f32x8 acc[4][2];
#pragma unroll
    for (int ms = 0; ms < 4; ++ms)
#pragma unroll
        for (int ns = 0; ns < 2; ++ns)
#pragma unroll
            for (int v = 0; v < 8; ++v) acc[ms][ns][v] = 0.f;

    gemm_mainloop(Comb, C2K_, W2, C2K_, C2K_, m0, n0, sm.At, sm.Bt, acc);

#pragma unroll
    for (int ms = 0; ms < 4; ++ms)
#pragma unroll
        for (int ns = 0; ns < 2; ++ns)
#pragma unroll
            for (int v = 0; v < 8; ++v) {
                int gr = m0 + wm * 64 + ms * 16 + v + 8 * hi;
                int gc = n0 + wn * 32 + ns * 16 + ln15;
                float val = acc[ms][ns][v] + b2[gc];
                size_t idx = (size_t)gr * DIM_ + gc;
                Out[idx] = X[idx] + gate[gc] * val;
            }
}

// --------------------------------------------------------------- launcher
extern "C" void kernel_launch(void* const* d_in, const int* in_sizes, int n_in,
                              void* d_out, int out_size, void* d_ws, size_t ws_size,
                              hipStream_t stream) {
    const float* x    = (const float*)d_in[0];
    const float* pe   = (const float*)d_in[1];
    const float* msc  = (const float*)d_in[2];
    const float* msh  = (const float*)d_in[3];
    const float* mgt  = (const float*)d_in[4];
    const float* mask = (const float*)d_in[5];
    const float* W1   = (const float*)d_in[6];
    const float* b1   = (const float*)d_in[7];
    const float* qnw  = (const float*)d_in[8];
    const float* knw  = (const float*)d_in[9];
    const float* W2   = (const float*)d_in[10];
    const float* b2   = (const float*)d_in[11];
    float* out = (float*)d_out;

    char* w = (char*)d_ws;
    size_t off = 0;
    __bf16* Xm   = (__bf16*)(w + off); off += (size_t)L_ * DIM_ * 2;        // 12.6 MB
    __bf16* Qb   = (__bf16*)(w + off); off += (size_t)H_ * L_ * HD_ * 2;    // 12.6 MB
    __bf16* Kb   = (__bf16*)(w + off); off += (size_t)H_ * L_ * HD_ * 2;    // 12.6 MB
    __bf16* Vb   = (__bf16*)(w + off); off += (size_t)H_ * HD_ * L_ * 2;    // 12.6 MB
    __bf16* Comb = (__bf16*)(w + off); off += (size_t)L_ * C2K_ * 2;        // 62.9 MB

    ln_mod_kernel<<<L_, 256, 0, stream>>>(x, msc, msh, Xm);
    gemm1_kernel<<<16 * (O1_ / 128), 256, 0, stream>>>(Xm, W1, b1, pe, qnw, knw,
                                                       Qb, Kb, Vb, Comb);
    attn_kernel<<<H_ * (L_ / 64), 128, 0, stream>>>(Qb, Kb, Vb, mask, Comb);
    gemm2_kernel<<<16 * (DIM_ / 128), 256, 0, stream>>>(Comb, W2, b2, x, mgt, out);
}
